// GNNModel_53970559041859
// MI455X (gfx1250) — compile-verified, matrix operand order checked by reading, not run on hardware
//
#include <hip/hip_runtime.h>

typedef __attribute__((ext_vector_type(16))) __bf16 v16bf;
typedef __attribute__((ext_vector_type(8)))  float  v8f;

#define HID 64

// ---------------- utility kernels ----------------
__global__ void zero_f32(float* __restrict__ p, long n) {
  long i = (long)blockIdx.x * blockDim.x + threadIdx.x;
  if (i < n) p[i] = 0.0f;
}

__global__ void degree_kernel(const int* __restrict__ dst, float* __restrict__ deg, int nE) {
  int e = blockIdx.x * blockDim.x + threadIdx.x;
  if (e < nE) atomicAdd(&deg[dst[e]], 1.0f);
}

__global__ void dinv_kernel(float* __restrict__ deg, int n) {
  int i = blockIdx.x * blockDim.x + threadIdx.x;
  if (i < n) deg[i] = rsqrtf(deg[i] + 1.0f);  // +1 for self loop
}

// ---------------- WMMA GEMM: C[M,64] = A[M,K] (fp32) x W[K,64] (fp32->bf16) ----------------
// Block: 256 threads = 8 waves; each wave computes 16 rows x 64 cols.
// W staged in LDS pre-swizzled into the 16-bit B-fragment layout:
//   frag (kstep s, ntile t), lane L (l=L&15, hi=L>>4) holds 16 contiguous bf16:
//   j=0..7  -> k = 32*s + 8*hi + j,        n = 16*t + l
//   j=8..15 -> k = 32*s + 16 + 8*hi + j-8, n = 16*t + l
extern __shared__ __bf16 smem[];

__global__ __launch_bounds__(256) void gemm_wmma_bf16(
    const float* __restrict__ A, const float* __restrict__ W,
    float* __restrict__ C, int M, int K) {
  const int tid = threadIdx.x;

  // Stage + swizzle W into LDS as bf16 fragments
  for (int idx = tid; idx < K * HID; idx += 256) {
    int k = idx >> 6, n = idx & 63;
    int s  = k >> 5, r = k & 31;
    int hi = (r >> 3) & 1;
    int j  = (r & 7) + ((r & 16) ? 8 : 0);
    int t  = n >> 4, l = n & 15;
    int L  = (hi << 4) | l;
    smem[(((s << 2) + t) * 32 + L) * 16 + j] = (__bf16)W[idx];
  }
  __syncthreads();

  const int wave = tid >> 5;
  const int lane = tid & 31;
  const int l    = lane & 15;
  const int hi   = lane >> 4;
  const int rowBase = blockIdx.x * 128 + wave * 16;
  if (rowBase >= M) return;              // wave-uniform (M % 16 == 0)
  const int row = rowBase + l;

  v8f acc0 = {}, acc1 = {}, acc2 = {}, acc3 = {};
  const float* ap = A + (size_t)row * K + hi * 8;

  for (int k0 = 0; k0 < K; k0 += 32) {
    // A fragment: fp32 -> bf16 in registers (16-bit A layout, ISA 7.12.2)
    float4 a0 = *(const float4*)(ap + k0);
    float4 a1 = *(const float4*)(ap + k0 + 4);
    float4 a2 = *(const float4*)(ap + k0 + 16);
    float4 a3 = *(const float4*)(ap + k0 + 20);
    if (k0 + 32 < K) __builtin_prefetch(ap + k0 + 32, 0, 1);

    v16bf af;
    af[0]  = (__bf16)a0.x; af[1]  = (__bf16)a0.y; af[2]  = (__bf16)a0.z; af[3]  = (__bf16)a0.w;
    af[4]  = (__bf16)a1.x; af[5]  = (__bf16)a1.y; af[6]  = (__bf16)a1.z; af[7]  = (__bf16)a1.w;
    af[8]  = (__bf16)a2.x; af[9]  = (__bf16)a2.y; af[10] = (__bf16)a2.z; af[11] = (__bf16)a2.w;
    af[12] = (__bf16)a3.x; af[13] = (__bf16)a3.y; af[14] = (__bf16)a3.z; af[15] = (__bf16)a3.w;

    const int s = k0 >> 5;
    v16bf b0 = *(const v16bf*)(smem + (((s << 2) + 0) * 32 + lane) * 16);
    v16bf b1 = *(const v16bf*)(smem + (((s << 2) + 1) * 32 + lane) * 16);
    v16bf b2 = *(const v16bf*)(smem + (((s << 2) + 2) * 32 + lane) * 16);
    v16bf b3 = *(const v16bf*)(smem + (((s << 2) + 3) * 32 + lane) * 16);

    acc0 = __builtin_amdgcn_wmma_f32_16x16x32_bf16(false, af, false, b0, (short)0, acc0, false, false);
    acc1 = __builtin_amdgcn_wmma_f32_16x16x32_bf16(false, af, false, b1, (short)0, acc1, false, false);
    acc2 = __builtin_amdgcn_wmma_f32_16x16x32_bf16(false, af, false, b2, (short)0, acc2, false, false);
    acc3 = __builtin_amdgcn_wmma_f32_16x16x32_bf16(false, af, false, b3, (short)0, acc3, false, false);
  }

  // D layout: VGPR r, lanes 0-15 -> (M=r, N=l); lanes 16-31 -> (M=8+r, N=l)
  for (int r = 0; r < 8; ++r) {
    float* crow = C + (size_t)(rowBase + hi * 8 + r) * HID + l;
    crow[0]  = acc0[r];
    crow[16] = acc1[r];
    crow[32] = acc2[r];
    crow[48] = acc3[r];
  }
}

// ---------------- message passing ----------------
// acc[i,c] = hpre[i,c] * dinv[i]^2 + bias[c]   (self-loop message + bias as initial value)
__global__ void init_self(const float* __restrict__ hpre, const float* __restrict__ dinv,
                          const float* __restrict__ bias, float* __restrict__ acc, int N) {
  long idx = (long)blockIdx.x * blockDim.x + threadIdx.x;
  if (idx >= (long)N * HID) return;
  int node = (int)(idx >> 6);
  int c = (int)(idx & 63);
  float di = dinv[node];
  acc[idx] = hpre[idx] * di * di + bias[c];
}

// One wave (32 lanes) per edge; each lane handles 2 channels.
__global__ void scatter_edges(const int* __restrict__ src, const int* __restrict__ dst,
                              const float* __restrict__ dinv, const float* __restrict__ hpre,
                              float* __restrict__ acc, int nE) {
  long idx = (long)blockIdx.x * blockDim.x + threadIdx.x;
  int e = (int)(idx >> 5);
  if (e >= nE) return;
  int c2 = ((int)idx & 31) * 2;
  int s = src[e], d = dst[e];
  float norm = dinv[s] * dinv[d];
  float2 v = *(const float2*)(hpre + (size_t)s * HID + c2);
  atomicAdd(&acc[(size_t)d * HID + c2 + 0], v.x * norm);
  atomicAdd(&acc[(size_t)d * HID + c2 + 1], v.y * norm);
}

__global__ void relu_kernel(float* __restrict__ p, long n) {
  long i = (long)blockIdx.x * blockDim.x + threadIdx.x;
  if (i < n) p[i] = fmaxf(p[i], 0.0f);
}

// ---------------- pooling + FC ----------------
__global__ void pool_accum(const float* __restrict__ h, const int* __restrict__ batch,
                           float* __restrict__ psum, float* __restrict__ pcnt, int N) {
  long idx = (long)blockIdx.x * blockDim.x + threadIdx.x;
  if (idx >= (long)N * HID) return;
  int node = (int)(idx >> 6);
  int c = (int)(idx & 63);
  int g = batch[node];
  atomicAdd(&psum[(size_t)g * HID + c], h[idx]);
  if (c == 0) atomicAdd(&pcnt[g], 1.0f);
}

__global__ void pool_finalize(const float* __restrict__ psum, const float* __restrict__ pcnt,
                              const float* __restrict__ Wfc, const float* __restrict__ bfc,
                              float* __restrict__ out) {
  int g = threadIdx.x;  // 256 graphs, one block
  float cnt = fmaxf(pcnt[g], 1.0f);
  float s = 0.0f;
#pragma unroll
  for (int c = 0; c < HID; ++c) s += psum[g * HID + c] * Wfc[c];
  out[g] = s / cnt + bfc[0];
}

// ---------------- host launcher ----------------
extern "C" void kernel_launch(void* const* d_in, const int* in_sizes, int n_in,
                              void* d_out, int out_size, void* d_ws, size_t ws_size,
                              hipStream_t stream) {
  const float* x    = (const float*)d_in[0];
  const int*   ei   = (const int*)  d_in[1];
  const int*   batch= (const int*)  d_in[2];
  const float* W1   = (const float*)d_in[3];
  const float* b1   = (const float*)d_in[4];
  const float* W2   = (const float*)d_in[5];
  const float* b2   = (const float*)d_in[6];
  const float* Wfc  = (const float*)d_in[7];
  const float* bfc  = (const float*)d_in[8];
  float* out = (float*)d_out;

  const int K1 = 512;
  const int N  = in_sizes[0] / K1;      // 100000 nodes
  const int nE = in_sizes[1] / 2;       // 1.6M edges
  const int G  = 256;

  const int* src = ei;
  const int* dst = ei + nE;

  // workspace layout
  float* ws   = (float*)d_ws;
  float* dinv = ws;                                   // N
  float* hpre = dinv + (((size_t)N + 255) & ~(size_t)255);
  float* acc  = hpre + (size_t)N * HID;
  float* psum = acc  + (size_t)N * HID;               // G*HID
  float* pcnt = psum + (size_t)G * HID;               // G

  const int T = 256;
  const long NH = (long)N * HID;

  // degrees -> dinv
  zero_f32<<<(N + T - 1) / T, T, 0, stream>>>(dinv, N);
  degree_kernel<<<(nE + T - 1) / T, T, 0, stream>>>(dst, dinv, nE);
  dinv_kernel<<<(N + T - 1) / T, T, 0, stream>>>(dinv, N);

  // ---- layer 1: h1 = relu(norm-agg(x @ W1) + b1) ----
  gemm_wmma_bf16<<<(N + 127) / 128, T, (size_t)K1 * HID * sizeof(__bf16), stream>>>(x, W1, hpre, N, K1);
  init_self<<<(NH + T - 1) / T, T, 0, stream>>>(hpre, dinv, b1, acc, N);
  scatter_edges<<<((long)nE * 32 + T - 1) / T, T, 0, stream>>>(src, dst, dinv, hpre, acc, nE);
  relu_kernel<<<(NH + T - 1) / T, T, 0, stream>>>(acc, NH);

  // ---- layer 2: h2 = relu(norm-agg(h1 @ W2) + b2) ----
  gemm_wmma_bf16<<<(N + 127) / 128, T, (size_t)HID * HID * sizeof(__bf16), stream>>>(acc, W2, hpre, N, HID);
  init_self<<<(NH + T - 1) / T, T, 0, stream>>>(hpre, dinv, b2, acc, N);
  scatter_edges<<<((long)nE * 32 + T - 1) / T, T, 0, stream>>>(src, dst, dinv, hpre, acc, nE);
  relu_kernel<<<(NH + T - 1) / T, T, 0, stream>>>(acc, NH);

  // ---- global mean pool + FC ----
  zero_f32<<<((long)G * HID + G + T - 1) / T, T, 0, stream>>>(psum, (long)G * HID + G);
  pool_accum<<<(NH + T - 1) / T, T, 0, stream>>>(acc, batch, psum, pcnt, N);
  pool_finalize<<<1, G, 0, stream>>>(psum, pcnt, Wfc, bfc, out);
}
